// GNN_28578712387708
// MI455X (gfx1250) — matrix operand order repaired and checked by reading, compiled, not run on hardware
//
#include <hip/hip_runtime.h>
#include <hip/hip_bf16.h>
#include <math.h>

typedef __attribute__((ext_vector_type(16))) __bf16 v16bf;
typedef __attribute__((ext_vector_type(8)))  float  v8f;

#define NB      1024
#define NNODE   32
#define FDIM    32
#define NNDIM   64
#define EEDGE   992
#define NTILES  62      // 992 / 16
#define NWAVES  8

// Branch-free tanh: native V_TANH_F32 if the builtin exists, else exp-based.
__device__ __forceinline__ float tanh_fast(float y) {
#if __has_builtin(__builtin_amdgcn_tanhf)
    return __builtin_amdgcn_tanhf(y);
#elif __has_builtin(__builtin_amdgcn_tanh_f32)
    return __builtin_amdgcn_tanh_f32(y);
#else
    float ay = __builtin_fabsf(y);
    float t  = __expf(-2.0f * ay);
    float th = 1.0f - 2.0f * t / (1.0f + t);   // tanh(|y|), branch-free
    return __builtin_copysignf(th, y);
#endif
}

// tanh-form GELU (branch-free, ~5 VALU + 1 trans vs libm erff's branchy ~30 ops)
__device__ __forceinline__ float gelu_fast(float x) {
    float x2 = x * x;
    float y  = x * __builtin_fmaf(x2, 0.0356774081f, 0.7978845608f); // 0.79788456*(x+0.044715x^3)
    float th = tanh_fast(y);
    return __builtin_fmaf(0.5f * x, th, 0.5f * x);
}
__device__ __forceinline__ float sigmoidf_(float x) {
    return 1.0f / (1.0f + __expf(-x));
}

// A fragment (bf16 16x32, MxK). row = pointer to this lane's matrix row (+K offset).
// Lane m = lane&15 is the row; half = lane>>4 selects K sub-pattern.
// VGPR v holds K = (v&3)*2 + (v>>2)*16 + half*8 and K+1 (contiguous -> b32/b128 DS loads).
__device__ __forceinline__ v16bf load_a_frag(const __bf16* row, int half) {
    v16bf a;
#pragma unroll
    for (int v = 0; v < 8; ++v) {
        int k = ((v & 3) << 1) + ((v >> 2) << 4) + (half << 3);
        a[2 * v]     = row[k];
        a[2 * v + 1] = row[k + 1];
    }
    return a;
}

// B fragment (bf16 32x16, KxN) from K-pair-interleaved LDS storage:
// element (k, n) lives at ((k>>1)*2*ld + 2*n + (k&1)).
// Lane n = lane&15 is the column; half = lane>>4; VGPR v holds K = ko + (v&3)*2 + (v>>2)*16 + half*8, K+1.
__device__ __forceinline__ v16bf load_b_frag(const __bf16* p, int ld, int ko, int no, int lane) {
    int n = (lane & 15) + no, half = lane >> 4;
    v16bf b;
#pragma unroll
    for (int v = 0; v < 8; ++v) {
        int k = ko + ((v & 3) << 1) + ((v >> 2) << 4) + (half << 3);
        const __bf16* q = p + (size_t)(k >> 1) * (2 * ld) + 2 * n;
        b[2 * v]     = q[0];
        b[2 * v + 1] = q[1];
    }
    return b;
}

#define WMMA_BF16(a, b, c) \
    __builtin_amdgcn_wmma_f32_16x16x32_bf16(false, (a), false, (b), (short)0, (c), false, false)

__global__ __launch_bounds__(256) void gnn_fused_kernel(
    const float* __restrict__ node0, const float* __restrict__ edge,
    const float* __restrict__ h0, const float* __restrict__ nmean,
    const float* __restrict__ nvari, const int* __restrict__ uis,
    const int* __restrict__ ujs, const int* __restrict__ itn,
    const float* __restrict__ W1a, const float* __restrict__ b1a,
    const float* __restrict__ W2a, const float* __restrict__ b2a,
    const float* __restrict__ W2b, const float* __restrict__ b2b,
    const float* __restrict__ W2c, const float* __restrict__ b2c,
    const float* __restrict__ Wih, const float* __restrict__ Whh,
    const float* __restrict__ bih, const float* __restrict__ bhh,
    const float* __restrict__ W3b, const float* __restrict__ b3b,
    const float* __restrict__ W4a, const float* __restrict__ b4a,
    const float* __restrict__ W4b, const float* __restrict__ b4b,
    const float* __restrict__ W4c, const float* __restrict__ b4c,
    const float* __restrict__ W5a, const float* __restrict__ b5a,
    const float* __restrict__ W5b, const float* __restrict__ b5b,
    float* __restrict__ out)
{
    // ---- LDS (total ~62.4 KB) ----
    __shared__ __align__(16) __bf16 sW2a[96 * 64];          // pair-interleaved, rows 66..95 zero
    __shared__ __align__(16) __bf16 sW2b[64 * 32];          // pair-interleaved
    __shared__ __align__(16) __bf16 sW2c[32 * 32];          // pair-interleaved
    __shared__ __align__(16) __bf16 sNodeB[32 * 32];        // node features, bf16 row-major
    __shared__ __align__(16) float  sNodeF[32 * 32];        // node features, f32
    __shared__ __align__(16) float  sH[32 * 64];            // GRU hidden state, f32
    __shared__ __align__(16) float  sMi[32 * 32];           // segment-sum accumulator
    __shared__ __align__(16) __bf16 sStage1[NWAVES * 16 * 64]; // per-wave m1 staging
    __shared__ __align__(16) __bf16 sStage2[NWAVES * 16 * 32]; // per-wave m2 staging
    __shared__ int   sUiTile[NWAVES * 16];                  // per-wave tile u_i cache
    __shared__ float sB2a[64];
    __shared__ float sB2b[32];
    __shared__ float sB2c[32];

    const int tid = threadIdx.x;
    const int b   = blockIdx.x;
    const int wave = tid >> 5, lane = tid & 31;
    const int mrow = lane & 15, half = lane >> 4;

    // ---- Init: weights -> LDS bf16 (W2a padded 66->96, K-pair interleaved) ----
    for (int idx = tid; idx < 96 * 64; idx += 256) {
        int r = idx >> 6, c = idx & 63;
        float v = (r < 66) ? W2a[r * 64 + c] : 0.0f;
        sW2a[(size_t)(r >> 1) * 128 + (c << 1) + (r & 1)] = (__bf16)v;
    }
    for (int idx = tid; idx < 64 * 32; idx += 256) {
        int r = idx >> 5, c = idx & 31;
        sW2b[(size_t)(r >> 1) * 64 + (c << 1) + (r & 1)] = (__bf16)W2b[idx];
    }
    for (int idx = tid; idx < 32 * 32; idx += 256) {
        int r = idx >> 5, c = idx & 31;
        sW2c[(size_t)(r >> 1) * 64 + (c << 1) + (r & 1)] = (__bf16)W2c[idx];
    }
    if (tid < 64) sB2a[tid] = b2a[tid];
    if (tid < 32) sB2b[tid] = b2b[tid];
    if (tid < 32) sB2c[tid] = b2c[tid];

    // ---- h init, node = node0 @ W1a + b1a ----
    for (int idx = tid; idx < 32 * 64; idx += 256) sH[idx] = h0[(size_t)b * 2048 + idx];
    for (int idx = tid; idx < 32 * 32; idx += 256) {
        int n = idx >> 5, f = idx & 31;
        float acc = b1a[f];
        for (int k = 0; k < 32; ++k)
            acc += node0[(size_t)b * 1024 + n * 32 + k] * W1a[k * 32 + f];
        sNodeF[idx] = acc;
        sNodeB[idx] = (__bf16)acc;
    }
    __syncthreads();

    const int iters = itn[0];
    for (int it = 0; it < iters; ++it) {
        // ---- zero segment-sum accumulator ----
        for (int idx = tid; idx < 32 * 32; idx += 256) sMi[idx] = 0.0f;
        __syncthreads();

        // ---- Edge MLP: each wave owns 16-edge tiles (WMMA bf16, f32 accum) ----
        {
            __bf16* st1 = sStage1 + (size_t)wave * (16 * 64);
            __bf16* st2 = sStage2 + (size_t)wave * (16 * 32);
            int*    sui = sUiTile + wave * 16;
            for (int tile = wave; tile < NTILES; tile += NWAVES) {
                const int e0  = tile * 16;
                const int myE = e0 + mrow;
                const int ri  = uis[myE];
                const int rj  = ujs[myE];
                if (half == 0) sui[mrow] = ri;   // cache this tile's u_i rows

                // A chunks of concat(node[u_i], node[u_j], [e0,e1,0..]) — never materialized
                v16bf a0 = load_a_frag(sNodeB + ri * 32, half);
                v16bf a1 = load_a_frag(sNodeB + rj * 32, half);
                const float* ep = edge + ((size_t)b * EEDGE + myE) * 2;
                float ev0 = ep[0], ev1 = ep[1];          // valid address for all lanes
                v16bf a2;
#pragma unroll
                for (int v = 0; v < 16; ++v) a2[v] = (__bf16)0.0f;
                a2[0] = (half == 0) ? (__bf16)ev0 : (__bf16)0.0f;  // select, no exec branch
                a2[1] = (half == 0) ? (__bf16)ev1 : (__bf16)0.0f;

                // Layer 1: [16x96] x [96x64] (K=66 zero-padded), GELU
#pragma unroll
                for (int nt = 0; nt < 4; ++nt) {
                    v8f acc = {};
                    acc = WMMA_BF16(a0, load_b_frag(sW2a, 64, 0,  nt * 16, lane), acc);
                    acc = WMMA_BF16(a1, load_b_frag(sW2a, 64, 32, nt * 16, lane), acc);
                    acc = WMMA_BF16(a2, load_b_frag(sW2a, 64, 64, nt * 16, lane), acc);
#pragma unroll
                    for (int r = 0; r < 8; ++r) {
                        int row = r + half * 8;
                        int col = nt * 16 + mrow;
                        st1[row * 64 + col] = (__bf16)gelu_fast(acc[r] + sB2a[col]);
                    }
                }
                // Layer 2: [16x64] x [64x32], GELU
                v16bf c0 = load_a_frag(st1 + mrow * 64,      half);
                v16bf c1 = load_a_frag(st1 + mrow * 64 + 32, half);
#pragma unroll
                for (int nt = 0; nt < 2; ++nt) {
                    v8f acc = {};
                    acc = WMMA_BF16(c0, load_b_frag(sW2b, 32, 0,  nt * 16, lane), acc);
                    acc = WMMA_BF16(c1, load_b_frag(sW2b, 32, 32, nt * 16, lane), acc);
#pragma unroll
                    for (int r = 0; r < 8; ++r) {
                        int row = r + half * 8;
                        int col = nt * 16 + mrow;
                        st2[row * 32 + col] = (__bf16)gelu_fast(acc[r] + sB2b[col]);
                    }
                }
                // Layer 3: [16x32] x [32x32], GELU, segment-sum into sMi (ds_add_f32)
                v16bf d0 = load_a_frag(st2 + mrow * 32, half);
#pragma unroll
                for (int nt = 0; nt < 2; ++nt) {
                    v8f acc = {};
                    acc = WMMA_BF16(d0, load_b_frag(sW2c, 32, 0, nt * 16, lane), acc);
#pragma unroll
                    for (int r = 0; r < 8; ++r) {
                        int row = r + half * 8;
                        int col = nt * 16 + mrow;
                        float mv = gelu_fast(acc[r] + sB2c[col]);
                        atomicAdd(&sMi[sui[row] * 32 + col], mv);
                    }
                }
            }
        }
        __syncthreads();

        // ---- GRU (scalar f32, c-major register blocking: 8 nodes/thread) ----
        {
            const int c  = tid & 63;
            const int n0 = (tid >> 6) * 8;
            float a_ir[8], a_iz[8], a_in[8], a_hr[8], a_hz[8], a_hn[8];
            float mk32[8], mk33[8];
#pragma unroll
            for (int j = 0; j < 8; ++j) {
                a_ir[j] = bih[c];       a_iz[j] = bih[64 + c];  a_in[j] = bih[128 + c];
                a_hr[j] = bhh[c];       a_hz[j] = bhh[64 + c];  a_hn[j] = bhh[128 + c];
                mk32[j] = nmean[(size_t)b * 32 + n0 + j];
                mk33[j] = 1.0f / nvari[(size_t)b * 32 + n0 + j];
            }
            for (int k = 0; k < 32; ++k) {
                float wr = Wih[k * 192 + c];
                float wz = Wih[k * 192 + 64 + c];
                float wn = Wih[k * 192 + 128 + c];
#pragma unroll
                for (int j = 0; j < 8; ++j) {
                    float mk = sMi[(n0 + j) * 32 + k];
                    a_ir[j] += mk * wr; a_iz[j] += mk * wz; a_in[j] += mk * wn;
                }
            }
#pragma unroll
            for (int j = 0; j < 8; ++j) {   // k = 32 (node_mean), k = 33 (1/vari)
                a_ir[j] += mk32[j] * Wih[32 * 192 + c];
                a_iz[j] += mk32[j] * Wih[32 * 192 + 64 + c];
                a_in[j] += mk32[j] * Wih[32 * 192 + 128 + c];
                a_ir[j] += mk33[j] * Wih[33 * 192 + c];
                a_iz[j] += mk33[j] * Wih[33 * 192 + 64 + c];
                a_in[j] += mk33[j] * Wih[33 * 192 + 128 + c];
            }
            for (int k = 0; k < 64; ++k) {
                float wr = Whh[k * 192 + c];
                float wz = Whh[k * 192 + 64 + c];
                float wn = Whh[k * 192 + 128 + c];
#pragma unroll
                for (int j = 0; j < 8; ++j) {
                    float hk = sH[(n0 + j) * 64 + k];
                    a_hr[j] += hk * wr; a_hz[j] += hk * wz; a_hn[j] += hk * wn;
                }
            }
            float hnew[8];
#pragma unroll
            for (int j = 0; j < 8; ++j) {
                int n = n0 + j;
                float r = sigmoidf_(a_ir[j] + a_hr[j]);
                float z = sigmoidf_(a_iz[j] + a_hz[j]);
                float g = gelu_fast(a_in[j] + r * a_hn[j]);
                hnew[j] = (1.0f - z) * g + z * sH[n * 64 + c];
            }
            __syncthreads();           // all reads of old sH complete
#pragma unroll
            for (int j = 0; j < 8; ++j) sH[(n0 + j) * 64 + c] = hnew[j];
        }
        __syncthreads();

        // ---- node = h @ W3b + b3b ----
        for (int idx = tid; idx < 32 * 32; idx += 256) {
            int n = idx >> 5, f = idx & 31;
            float acc = b3b[f];
            for (int k = 0; k < 64; ++k) acc += sH[n * 64 + k] * W3b[k * 32 + f];
            sNodeF[idx] = acc;
            sNodeB[idx] = (__bf16)acc;
        }
        __syncthreads();
    }

    // ---- Readout chain (scalar f32, staging buffers reuse LDS) ----
    float* t1 = (float*)sStage1;   // 32x64
    for (int idx = tid; idx < 32 * 64; idx += 256) {
        int n = idx >> 6, c = idx & 63;
        float acc = b4a[c];
        for (int k = 0; k < 32; ++k) acc += sNodeF[n * 32 + k] * W4a[k * 64 + c];
        t1[idx] = acc;
    }
    __syncthreads();
    for (int idx = tid; idx < 32 * 32; idx += 256) {      // t2 -> sMi
        int n = idx >> 5, c = idx & 31;
        float acc = b4b[c];
        for (int k = 0; k < 64; ++k) acc += t1[n * 64 + k] * W4b[k * 32 + c];
        sMi[idx] = acc;
    }
    __syncthreads();
    for (int idx = tid; idx < 32 * 32; idx += 256) {      // t3 -> sNodeF
        int n = idx >> 5, c = idx & 31;
        float acc = b4c[c];
        for (int k = 0; k < 32; ++k) acc += sMi[n * 32 + k] * W4c[k * 32 + c];
        sNodeF[idx] = acc;
    }
    __syncthreads();
    float* t4 = (float*)sStage2;   // 32x16
    for (int idx = tid; idx < 32 * 16; idx += 256) {
        int n = idx >> 4, c = idx & 15;
        float acc = b5a[c];
        for (int k = 0; k < 32; ++k) acc += sNodeF[n * 32 + k] * W5a[k * 16 + c];
        t4[idx] = acc;
    }
    __syncthreads();
    for (int idx = tid; idx < 32 * 2; idx += 256) {
        int n = idx >> 1, c = idx & 1;
        float acc = b5b[c];
        for (int k = 0; k < 16; ++k) acc += t4[n * 16 + k] * W5b[k * 2 + c];
        if (c == 0) out[(size_t)b * 32 + n] = acc;
        else        out[(size_t)NB * 32 + (size_t)b * 32 + n] = 1.0f / acc;
    }
}

extern "C" void kernel_launch(void* const* d_in, const int* in_sizes, int n_in,
                              void* d_out, int out_size, void* d_ws, size_t ws_size,
                              hipStream_t stream) {
    (void)in_sizes; (void)n_in; (void)out_size; (void)d_ws; (void)ws_size;
    const float* node0 = (const float*)d_in[0];
    const float* edge  = (const float*)d_in[1];
    const float* h0    = (const float*)d_in[2];
    const float* nmean = (const float*)d_in[3];
    const float* nvari = (const float*)d_in[4];
    const int*   uis   = (const int*)d_in[5];
    const int*   ujs   = (const int*)d_in[6];
    const int*   itn   = (const int*)d_in[7];
    const float* W1a = (const float*)d_in[8];  const float* b1a = (const float*)d_in[9];
    const float* W2a = (const float*)d_in[10]; const float* b2a = (const float*)d_in[11];
    const float* W2b = (const float*)d_in[12]; const float* b2b = (const float*)d_in[13];
    const float* W2c = (const float*)d_in[14]; const float* b2c = (const float*)d_in[15];
    const float* Wih = (const float*)d_in[16]; const float* Whh = (const float*)d_in[17];
    const float* bih = (const float*)d_in[18]; const float* bhh = (const float*)d_in[19];
    const float* W3b = (const float*)d_in[20]; const float* b3b = (const float*)d_in[21];
    const float* W4a = (const float*)d_in[22]; const float* b4a = (const float*)d_in[23];
    const float* W4b = (const float*)d_in[24]; const float* b4b = (const float*)d_in[25];
    const float* W4c = (const float*)d_in[26]; const float* b4c = (const float*)d_in[27];
    const float* W5a = (const float*)d_in[28]; const float* b5a = (const float*)d_in[29];
    const float* W5b = (const float*)d_in[30]; const float* b5b = (const float*)d_in[31];

    gnn_fused_kernel<<<dim3(NB), dim3(256), 0, stream>>>(
        node0, edge, h0, nmean, nvari, uis, ujs, itn,
        W1a, b1a, W2a, b2a, W2b, b2b, W2c, b2c,
        Wih, Whh, bih, bhh, W3b, b3b,
        W4a, b4a, W4b, b4b, W4c, b4c, W5a, b5a, W5b, b5b,
        (float*)d_out);
}
